// Net_22041772163090
// MI455X (gfx1250) — compile-verified, matrix operand order checked by reading, not run on hardware
//
#include <hip/hip_runtime.h>
#include <hip/hip_bf16.h>
#include <math.h>

// ---------------------------------------------------------------------------
// 2-layer GAT for MI455X (gfx1250, wave32).
// GEMM1: 64x512 block tile; A panel async-staged to LDS once per block
//        (global_load_async_to_lds_b128 + s_wait_asynccnt), conflict-free
//        ds_load_b128 fragment reads; v_wmma_f32_16x16x32_bf16, f32 accum.
// Edge softmax: ordered-uint atomicMax + f32 atomicAdd (L2-resident).
// ---------------------------------------------------------------------------

typedef __attribute__((ext_vector_type(16))) __bf16   v16bf;
typedef __attribute__((ext_vector_type(8)))  float    v8f;
typedef __attribute__((ext_vector_type(8)))  unsigned v8u;

#define N_NODES   20000
#define NUM_FEAT  512
#define HC        512      // HEADS*HIDDEN
#define HEADS     8
#define HIDDEN    64
#define NCLS      32
#define NEG_SLOPE 0.2f

// LDS row stride for the A panel: 512*2 data bytes + 16 pad bytes.
// 1040/4 = 260 ≡ 4 (mod 64) -> the 16 lanes of a fragment hit disjoint
// 4-bank groups covering all 64 banks: conflict-free ds_load_b128.
#define A_LDS_STRIDE 1040

__device__ __forceinline__ unsigned short f2bf(float f) {
  unsigned u = __float_as_uint(f);
  u += 0x7fffu + ((u >> 16) & 1u);            // round-to-nearest-even
  return (unsigned short)(u >> 16);
}
// order-preserving float <-> uint for atomicMax on signed floats
__device__ __forceinline__ unsigned enc_f(float f) {
  unsigned u = __float_as_uint(f);
  return (u & 0x80000000u) ? ~u : (u | 0x80000000u);
}
__device__ __forceinline__ float dec_f(unsigned u) {
  unsigned v = (u & 0x80000000u) ? (u & 0x7fffffffu) : ~u;
  return __uint_as_float(v);
}

__device__ __forceinline__ v16bf frag_from(uint4 lo, uint4 hi) {
  v8u u;
  u[0] = lo.x; u[1] = lo.y; u[2] = lo.z; u[3] = lo.w;
  u[4] = hi.x; u[5] = hi.y; u[6] = hi.z; u[7] = hi.w;
  return __builtin_bit_cast(v16bf, u);
}

// ---------------- casts / packing --------------------------------------------
__global__ void k_cast_x(const float* __restrict__ x, unsigned short* __restrict__ xb,
                         long n_valid, long n_total) {
  long i = (long)blockIdx.x * blockDim.x + threadIdx.x;
  if (i >= n_total) return;
  xb[i] = (i < n_valid) ? f2bf(x[i]) : (unsigned short)0;
}

// Pack W[K x Ncols] (row-major, stride ncols) into WMMA B-fragment order:
// uint index = (((n*16 + kt)*2 + g)*8 + p), holding bf16 pair at
// k = kt*32 + g*8 + (p<4 ? 2p : 2p+8)  (and k+1).
__global__ void k_pack_w(const float* __restrict__ W, unsigned* __restrict__ wf,
                         int ncols, long n) {
  long i = (long)blockIdx.x * blockDim.x + threadIdx.x;
  if (i >= n) return;
  int p  = (int)(i & 7);
  int g  = (int)((i >> 3) & 1);
  int kt = (int)((i >> 4) & 15);
  long nn = i >> 8;
  int k = kt * 32 + g * 8 + ((p < 4) ? 2 * p : 2 * p + 8);
  unsigned lo = f2bf(W[(long)k * ncols + nn]);
  unsigned hi = f2bf(W[(long)(k + 1) * ncols + nn]);
  wf[i] = lo | (hi << 16);
}

__global__ void k_fill_u32(unsigned* __restrict__ p, unsigned v, long n) {
  long i = (long)blockIdx.x * blockDim.x + threadIdx.x;
  if (i < n) p[i] = v;
}

// ---------------- GEMM1: h1[N,512] = xb @ W1  (bf16 WMMA, f32 accum) --------
// Block = 256 threads = 8 waves; block tile 64(M) x 512(N); wave tile 64x64.
// A panel (64x512 bf16) async-staged into LDS once, read 8x from LDS.
__global__ void __launch_bounds__(256)
k_gemm1(const unsigned short* __restrict__ xb, const unsigned* __restrict__ w1f,
        float* __restrict__ h1, int nrows) {
  __shared__ __align__(16) unsigned char asmem[64 * A_LDS_STRIDE];

  const int lane  = threadIdx.x & 31;
  const int wave  = threadIdx.x >> 5;
  const int l15   = lane & 15;
  const int g     = lane >> 4;
  const int khalf = g * 8;
  const int nbase = wave * 64;
  const long rowbase = (long)blockIdx.x * 64;

  // ---- async-stage A panel: 4096 x 16B chunks, 16 per thread --------------
  {
    const unsigned short* gbase = xb + rowbase * NUM_FEAT;
#pragma unroll 4
    for (int i = 0; i < 16; ++i) {
      int chunk = (int)threadIdx.x + i * 256;
      int row = chunk >> 6, c16 = chunk & 63;
      unsigned long long ga =
          (unsigned long long)(const void*)(gbase + (size_t)row * NUM_FEAT + c16 * 8);
      unsigned la =
          (unsigned)(unsigned long long)(const void*)(asmem + row * A_LDS_STRIDE + c16 * 16);
      asm volatile("global_load_async_to_lds_b128 %0, %1, off"
                   :: "v"(la), "v"(ga) : "memory");
    }
    asm volatile("s_wait_asynccnt 0x0" ::: "memory");
    __syncthreads();
  }

  v8f acc[4][4] = {};
  for (int kt = 0; kt < 16; ++kt) {
    const int kb = kt * 32;
    // A fragments from LDS (conflict-free b128 reads)
    v16bf afrag[4];
#pragma unroll
    for (int ms = 0; ms < 4; ++ms) {
      const unsigned char* ap =
          asmem + (ms * 16 + l15) * A_LDS_STRIDE + (kb + khalf) * 2;
      afrag[ms] = frag_from(*(const uint4*)(ap), *(const uint4*)(ap + 32));
    }
    // B fragments from global (W1 is L2-resident), reused across 4 M-subtiles
#pragma unroll
    for (int ns = 0; ns < 4; ++ns) {
      const unsigned* bp =
          w1f + ((((size_t)(nbase + ns * 16 + l15)) * 16 + kt) * 2 + g) * 8;
      v16bf bfrag = frag_from(*(const uint4*)(bp), *(const uint4*)(bp + 4));
#pragma unroll
      for (int ms = 0; ms < 4; ++ms)
        acc[ms][ns] = __builtin_amdgcn_wmma_f32_16x16x32_bf16(
            false, afrag[ms], false, bfrag, (short)0, acc[ms][ns], false, false);
    }
  }

#pragma unroll
  for (int ms = 0; ms < 4; ++ms) {
#pragma unroll
    for (int ns = 0; ns < 4; ++ns) {
      const int col = nbase + ns * 16 + l15;
#pragma unroll
      for (int i = 0; i < 8; ++i) {
        long m = rowbase + ms * 16 + i + g * 8;
        if (m < nrows) h1[m * HC + col] = acc[ms][ns][i];
      }
    }
  }
}

// ---------------- GEMM2: g2[N,32] = h2b @ W2 --------------------------------
// block = 256 = 8 waves; wave owns one 16-row tile, all 32 cols (2 subtiles).
__global__ void __launch_bounds__(256)
k_gemm2(const unsigned short* __restrict__ hb, const unsigned* __restrict__ w2f,
        float* __restrict__ g2, int nrows, int ntiles) {
  const int lane = threadIdx.x & 31;
  const int wave = threadIdx.x >> 5;
  const int tile = blockIdx.x * 8 + wave;
  if (tile >= ntiles) return;                    // wave-uniform: EXEC stays full
  const int l15 = lane & 15, g = lane >> 4, khalf = g * 8;
  const unsigned short* arow = hb + ((size_t)tile * 16 + l15) * (size_t)HC;

  v8f acc[2] = {};
  for (int kt = 0; kt < 16; ++kt) {
    const int kb = kt * 32;
    v16bf a = frag_from(*(const uint4*)(arow + kb + khalf),
                        *(const uint4*)(arow + kb + khalf + 16));
#pragma unroll
    for (int s = 0; s < 2; ++s) {
      const unsigned* bp = w2f + ((((size_t)(s * 16 + l15)) * 16 + kt) * 2 + g) * 8;
      v16bf b = frag_from(*(const uint4*)(bp), *(const uint4*)(bp + 4));
      acc[s] = __builtin_amdgcn_wmma_f32_16x16x32_bf16(false, a, false, b,
                                                       (short)0, acc[s], false, false);
    }
  }
#pragma unroll
  for (int s = 0; s < 2; ++s) {
    const int col = s * 16 + l15;
#pragma unroll
    for (int i = 0; i < 8; ++i) {
      long m = (long)tile * 16 + i + g * 8;
      if (m < nrows) g2[m * NCLS + col] = acc[s][i];
    }
  }
}

// ---------------- attention logits ------------------------------------------
__global__ void k_alpha1(const float* __restrict__ h1, const float* __restrict__ asw,
                         const float* __restrict__ adw, float* __restrict__ as_,
                         float* __restrict__ ad_, int n) {
  int t = blockIdx.x * blockDim.x + threadIdx.x;
  if (t >= n) return;                            // n = N*HEADS
  int node = t >> 3, h = t & 7;
  const float* hp = h1 + (size_t)node * HC + h * HIDDEN;
  const float* ps = asw + h * HIDDEN;
  const float* pd = adw + h * HIDDEN;
  float ss = 0.f, sd = 0.f;
#pragma unroll 8
  for (int c = 0; c < HIDDEN; ++c) { float v = hp[c]; ss += v * ps[c]; sd += v * pd[c]; }
  as_[t] = ss; ad_[t] = sd;
}

__global__ void k_alpha2(const float* __restrict__ g2, const float* __restrict__ asw,
                         const float* __restrict__ adw, float* __restrict__ as_,
                         float* __restrict__ ad_, int n) {
  int t = blockIdx.x * blockDim.x + threadIdx.x;
  if (t >= n) return;
  const float* gp = g2 + (size_t)t * NCLS;
  float ss = 0.f, sd = 0.f;
#pragma unroll
  for (int c = 0; c < NCLS; ++c) { float v = gp[c]; ss += v * asw[c]; sd += v * adw[c]; }
  as_[t] = ss; ad_[t] = sd;
}

// ---------------- edge softmax layer 1 (H=8) --------------------------------
__device__ __forceinline__ void edge_sd(const int* ei, int E, int e, int& s, int& d) {
  if (e < E) { s = ei[e]; d = ei[E + e]; } else { s = d = e - E; }
}

__global__ void k_edge_max1(const int* __restrict__ ei, int E, int ET8,
                            const float* __restrict__ as_, const float* __restrict__ ad_,
                            unsigned* __restrict__ m) {
  int t = blockIdx.x * blockDim.x + threadIdx.x;
  if (t >= ET8) return;
  int e = t >> 3, h = t & 7, s, d;
  edge_sd(ei, E, e, s, d);
  float v = as_[s * HEADS + h] + ad_[d * HEADS + h];
  v = v > 0.f ? v : NEG_SLOPE * v;
  atomicMax(m + d * HEADS + h, enc_f(v));
}

__global__ void k_edge_exp1(const int* __restrict__ ei, int E, int ET8,
                            const float* __restrict__ as_, const float* __restrict__ ad_,
                            const unsigned* __restrict__ m, float* __restrict__ e1,
                            float* __restrict__ dn) {
  int t = blockIdx.x * blockDim.x + threadIdx.x;
  if (t >= ET8) return;
  int e = t >> 3, h = t & 7, s, d;
  edge_sd(ei, E, e, s, d);
  float v = as_[s * HEADS + h] + ad_[d * HEADS + h];
  v = v > 0.f ? v : NEG_SLOPE * v;
  float ex = expf(v - dec_f(m[d * HEADS + h]));
  e1[t] = ex;
  atomicAdd(dn + d * HEADS + h, ex);
}

// aggregation: one block per edge, 256 threads, 2 channels each (512 total)
__global__ void __launch_bounds__(256)
k_agg1(const int* __restrict__ ei, int E, const float* __restrict__ h1,
       const float* __restrict__ e1, const float* __restrict__ dn,
       float* __restrict__ out) {
  int e = blockIdx.x, s, d;
  edge_sd(ei, E, e, s, d);
#pragma unroll
  for (int r = 0; r < 2; ++r) {
    int c = threadIdx.x + r * 256;
    int h = c >> 6;
    float alpha = e1[(size_t)e * HEADS + h] / (dn[(size_t)d * HEADS + h] + 1e-16f);
    atomicAdd(out + (size_t)d * HC + c, h1[(size_t)s * HC + c] * alpha);
  }
}

// ---------------- ELU + bias + cast to bf16 ---------------------------------
__global__ void k_elu_cast(const float* __restrict__ out1, const float* __restrict__ b1,
                           unsigned short* __restrict__ hb, long n_valid, long n_total) {
  long i = (long)blockIdx.x * blockDim.x + threadIdx.x;
  if (i >= n_total) return;
  float v = 0.f;
  if (i < n_valid) {
    v = out1[i] + b1[i & (HC - 1)];
    v = v > 0.f ? v : (expf(v) - 1.f);
  }
  hb[i] = f2bf(v);
}

// ---------------- edge softmax layer 2 (H=1) --------------------------------
__global__ void k_edge_max2(const int* __restrict__ ei, int E, int ET,
                            const float* __restrict__ as_, const float* __restrict__ ad_,
                            unsigned* __restrict__ m) {
  int e = blockIdx.x * blockDim.x + threadIdx.x;
  if (e >= ET) return;
  int s, d; edge_sd(ei, E, e, s, d);
  float v = as_[s] + ad_[d];
  v = v > 0.f ? v : NEG_SLOPE * v;
  atomicMax(m + d, enc_f(v));
}

__global__ void k_edge_exp2(const int* __restrict__ ei, int E, int ET,
                            const float* __restrict__ as_, const float* __restrict__ ad_,
                            const unsigned* __restrict__ m, float* __restrict__ e2,
                            float* __restrict__ dn) {
  int e = blockIdx.x * blockDim.x + threadIdx.x;
  if (e >= ET) return;
  int s, d; edge_sd(ei, E, e, s, d);
  float v = as_[s] + ad_[d];
  v = v > 0.f ? v : NEG_SLOPE * v;
  float ex = expf(v - dec_f(m[d]));
  e2[e] = ex;
  atomicAdd(dn + d, ex);
}

__global__ void k_agg2(const int* __restrict__ ei, int E, long ET32,
                       const float* __restrict__ g2, const float* __restrict__ e2,
                       const float* __restrict__ dn, float* __restrict__ out) {
  long t = (long)blockIdx.x * blockDim.x + threadIdx.x;
  if (t >= ET32) return;
  int e = (int)(t >> 5), c = (int)(t & 31), s, d;
  edge_sd(ei, E, e, s, d);
  float alpha = e2[e] / (dn[d] + 1e-16f);
  atomicAdd(out + (size_t)d * NCLS + c, g2[(size_t)s * NCLS + c] * alpha);
}

// ---------------- log_softmax ------------------------------------------------
__global__ void k_lsm(const float* __restrict__ acc, const float* __restrict__ b2,
                      float* __restrict__ out, int N) {
  int n = blockIdx.x * blockDim.x + threadIdx.x;
  if (n >= N) return;
  float v[NCLS], mx = -INFINITY;
#pragma unroll
  for (int c = 0; c < NCLS; ++c) { v[c] = acc[(size_t)n * NCLS + c] + b2[c]; mx = fmaxf(mx, v[c]); }
  float s = 0.f;
#pragma unroll
  for (int c = 0; c < NCLS; ++c) s += expf(v[c] - mx);
  float l = logf(s) + mx;
#pragma unroll
  for (int c = 0; c < NCLS; ++c) out[(size_t)n * NCLS + c] = v[c] - l;
}

// ---------------------------------------------------------------------------
static inline unsigned g1d(long n, int b = 256) { return (unsigned)((n + b - 1) / b); }

extern "C" void kernel_launch(void* const* d_in, const int* in_sizes, int n_in,
                              void* d_out, int out_size, void* d_ws, size_t ws_size,
                              hipStream_t stream) {
  const float* x    = (const float*)d_in[0];
  const int*   ei   = (const int*)  d_in[1];
  const float* W1   = (const float*)d_in[2];
  const float* as1w = (const float*)d_in[3];
  const float* ad1w = (const float*)d_in[4];
  const float* b1   = (const float*)d_in[5];
  const float* W2   = (const float*)d_in[6];
  const float* as2w = (const float*)d_in[7];
  const float* ad2w = (const float*)d_in[8];
  const float* b2   = (const float*)d_in[9];
  float* out = (float*)d_out;

  const int N  = in_sizes[0] / NUM_FEAT;          // 20000
  const int E  = in_sizes[1] / 2;                 // 320000
  const int ET = E + N;                           // edges + self loops
  const int MB   = (N + 63) / 64;                 // gemm1 64-row blocks
  const int NP64 = MB * 64;                       // xb padded rows
  const int MT   = (N + 15) / 16;                 // gemm2 16-row tiles
  const int NP16 = MT * 16;                       // h2b padded rows

  // --- carve workspace (~145 MB required) ---------------------------------
  char* ws = (char*)d_ws;
  auto carve = [&](size_t bytes) -> char* {
    char* p = ws; ws += (bytes + 255) & ~(size_t)255; return p;
  };
  unsigned short* xb  = (unsigned short*)carve((size_t)NP64 * NUM_FEAT * 2);
  unsigned*       w1f = (unsigned*)      carve((size_t)HC * NUM_FEAT * 2);
  float*          h1  = (float*)         carve((size_t)N * HC * 4);
  float*          as1 = (float*)         carve((size_t)N * HEADS * 4);
  float*          ad1 = (float*)         carve((size_t)N * HEADS * 4);
  unsigned*       m1  = (unsigned*)      carve((size_t)N * HEADS * 4);
  float*          dn1 = (float*)         carve((size_t)N * HEADS * 4);
  float*          e1  = (float*)         carve((size_t)ET * HEADS * 4);
  float*          o1  = (float*)         carve((size_t)N * HC * 4);
  unsigned short* h2b = (unsigned short*)carve((size_t)NP16 * HC * 2);
  unsigned*       w2f = (unsigned*)      carve((size_t)NCLS * HC * 2);
  float*          g2  = (float*)         carve((size_t)N * NCLS * 4);
  float*          as2 = (float*)         carve((size_t)N * 4);
  float*          ad2 = (float*)         carve((size_t)N * 4);
  unsigned*       m2  = (unsigned*)      carve((size_t)N * 4);
  float*          dn2 = (float*)         carve((size_t)N * 4);
  float*          e2  = (float*)         carve((size_t)ET * 4);
  float*          o2  = (float*)         carve((size_t)N * NCLS * 4);
  (void)ws_size; (void)n_in; (void)out_size;

  // --- layer 1 -------------------------------------------------------------
  k_cast_x<<<g1d((long)NP64 * NUM_FEAT), 256, 0, stream>>>(x, xb, (long)N * NUM_FEAT,
                                                           (long)NP64 * NUM_FEAT);
  k_pack_w<<<g1d((long)HC * NUM_FEAT / 2), 256, 0, stream>>>(W1, w1f, HC,
                                                             (long)HC * NUM_FEAT / 2);
  k_pack_w<<<g1d((long)NCLS * HC / 2), 256, 0, stream>>>(W2, w2f, NCLS,
                                                         (long)NCLS * HC / 2);
  k_gemm1<<<MB, 256, 0, stream>>>(xb, w1f, h1, N);
  k_alpha1<<<g1d((long)N * HEADS), 256, 0, stream>>>(h1, as1w, ad1w, as1, ad1, N * HEADS);

  k_fill_u32<<<g1d((long)N * HEADS), 256, 0, stream>>>(m1, 0u, (long)N * HEADS);
  k_fill_u32<<<g1d((long)N * HEADS), 256, 0, stream>>>((unsigned*)dn1, 0u, (long)N * HEADS);
  k_fill_u32<<<g1d((long)N * HC), 256, 0, stream>>>((unsigned*)o1, 0u, (long)N * HC);

  k_edge_max1<<<g1d((long)ET * HEADS), 256, 0, stream>>>(ei, E, ET * HEADS, as1, ad1, m1);
  k_edge_exp1<<<g1d((long)ET * HEADS), 256, 0, stream>>>(ei, E, ET * HEADS, as1, ad1, m1, e1, dn1);
  k_agg1<<<ET, 256, 0, stream>>>(ei, E, h1, e1, dn1, o1);

  k_elu_cast<<<g1d((long)NP16 * HC), 256, 0, stream>>>(o1, b1, h2b, (long)N * HC,
                                                       (long)NP16 * HC);

  // --- layer 2 -------------------------------------------------------------
  k_gemm2<<<g1d(MT, 8), 256, 0, stream>>>(h2b, w2f, g2, N, MT);
  k_alpha2<<<g1d(N), 256, 0, stream>>>(g2, as2w, ad2w, as2, ad2, N);

  k_fill_u32<<<g1d(N), 256, 0, stream>>>(m2, 0u, N);
  k_fill_u32<<<g1d(N), 256, 0, stream>>>((unsigned*)dn2, 0u, N);
  k_fill_u32<<<g1d((long)N * NCLS), 256, 0, stream>>>((unsigned*)o2, 0u, (long)N * NCLS);

  k_edge_max2<<<g1d(ET), 256, 0, stream>>>(ei, E, ET, as2, ad2, m2);
  k_edge_exp2<<<g1d(ET), 256, 0, stream>>>(ei, E, ET, as2, ad2, m2, e2, dn2);
  k_agg2<<<g1d((long)ET * NCLS), 256, 0, stream>>>(ei, E, (long)ET * NCLS, g2, e2, dn2, o2);

  k_lsm<<<g1d(N), 256, 0, stream>>>(o2, b2, out, N);
}
